// WeightMergeNet_84318797955666
// MI455X (gfx1250) — matrix operand order repaired and checked by reading, not compiled, not run on hardware
//
#include <hip/hip_runtime.h>
#include <cstdint>

typedef __attribute__((ext_vector_type(16))) _Float16 v16h;
typedef __attribute__((ext_vector_type(8)))  float    v8f;

#define KBOX 16
#define DDIM 7
#define XSTR 116                       // padded dwords per group in LDS (112 used, 16B-aligned)
#define CSTR 17                        // padded coef stride (odd -> conflict-free)
#define WAVE_LDS (32*XSTR + 32*CSTR)   // 4256 dwords per wave (17024B, 16B multiple)

__global__ void __launch_bounds__(64)
weight_merge_kernel(const float* __restrict__ x,
                    const float* __restrict__ conf,
                    const float* __restrict__ w,
                    float* __restrict__ out, int G)
{
  __shared__ __align__(16) float smem[2 * WAVE_LDS];
  const int  lane  = threadIdx.x & 31;
  const int  wib   = threadIdx.x >> 5;
  const long gwave = (long)blockIdx.x * 2 + wib;
  const long Gbase = gwave * 32;
  if (Gbase >= (long)G) return;

  float* xs = smem + wib * WAVE_LDS;   // 32 groups x 116 dwords of x (f32)
  float* cs = xs + 32 * XSTR;          // 32 groups x 17 dwords of coef (f32)

  if (Gbase + 32 <= (long)G) {
    // ---------------- Phase 0: async stage x -> LDS (coalesced 16B chunks) ----------------
    const float*   gsrc  = x + Gbase * (KBOX * DDIM);
    const uint32_t xsOff = (uint32_t)(uintptr_t)xs;
#pragma unroll
    for (int j = 0; j < 28; ++j) {
      int c = j * 32 + lane;                      // contiguous chunk id over 32*112 dwords
      int g = c / 28;
      int i = c - g * 28;
      uint32_t     dst = xsOff + (uint32_t)(g * XSTR + i * 4) * 4u;
      const float* src = gsrc + c * 4;
      asm volatile("global_load_async_to_lds_b128 %0, %1, off"
                   :: "v"(dst), "v"(src) : "memory");
    }

    // overlap async copy with per-lane conf row + weights
    float cv[16], wv[16];
    const float* cp = conf + (Gbase + lane) * KBOX;
#pragma unroll
    for (int i = 0; i < 4; ++i) ((float4*)cv)[i] = ((const float4*)cp)[i];
#pragma unroll
    for (int i = 0; i < 4; ++i) ((float4*)wv)[i] = ((const float4*)w)[i];

    asm volatile("s_wait_asynccnt 0x0" ::: "memory");

    // ---------------- Phase 1: per-group scalars -> combined coefficient ----------------
    {
      const float* myx = xs + lane * XSTR;
      int kcnt = 0;
#pragma unroll
      for (int k = 0; k < KBOX; ++k) {
        const float* r = myx + k * DDIM;
        int nz = (r[0] != 0.f) | (r[1] != 0.f) | (r[2] != 0.f) | (r[3] != 0.f) |
                 (r[4] != 0.f) | (r[5] != 0.f) | (r[6] != 0.f);
        kcnt += nz;
      }
      float wsum = 0.f, csum = 0.f;
#pragma unroll
      for (int k = 0; k < KBOX; ++k) if (k < kcnt) { wsum += wv[k]; csum += cv[k]; }
      const float iw = 0.5f / wsum, ic = 0.5f / csum;
#pragma unroll
      for (int k = 0; k < KBOX; ++k) cs[lane * CSTR + k] = wv[k] * iw + cv[k] * ic;
    }

    // ---------------- Phase 2: block-diagonal WMMA contraction ----------------
    const int m = lane & 15;    // N index (= d), also A row within tile
    const int h = lane >> 4;    // lane half
#pragma unroll
    for (int tb = 0; tb < 32; tb += 16) {          // two 16-group tiles per wave
      _Float16 chh[8], cll[8];                     // coef hi/lo halves, A-layout K = e + 8h
#pragma unroll
      for (int e = 0; e < 8; ++e) {
        float    c  = cs[(tb + m) * CSTR + e + 8 * h];
        _Float16 hi = (_Float16)c;
        chh[e] = hi;
        cll[e] = (_Float16)(c - (float)hi);
      }
      v8f acc = {0.f, 0.f, 0.f, 0.f, 0.f, 0.f, 0.f, 0.f};
#pragma unroll
      for (int t = 0; t < 8; ++t) {                // contraction chunk: groups 2t, 2t+1
        const int    grel = tb + 2 * t;
        // Lanes m>=7 gather "garbage" columns; those feed only D columns 7..15,
        // which are never stored, so no predication/zeroing is needed. All
        // addresses stay inside this wave's smem slice (max dword 3716 < 4256).
        const float* xb = xs + grel * XSTR + 8 * h * DDIM + m;
        v16h Bh, Bl;                               // B element e = x[grel + (e>>3), 8h + (e&7), n]
#pragma unroll
        for (int k = 0; k < 16; ++k) {
          float    xv = xb[(k >> 3) * XSTR + (k & 7) * DDIM];
          _Float16 hi = (_Float16)xv;
          Bh[k] = hi;
          Bl[k] = (_Float16)(xv - (float)hi);
        }
        const bool selLo = (m == 2 * t);           // this lane's row is group 2t
        const bool selHi = (m == 2 * t + 1);       // this lane's row is group 2t+1
        v16h Ah, Al;
#pragma unroll
        for (int e = 0; e < 8; ++e) {
          Ah[e]     = selLo ? chh[e] : (_Float16)0.f;
          Ah[e + 8] = selHi ? chh[e] : (_Float16)0.f;
          Al[e]     = selLo ? cll[e] : (_Float16)0.f;
          Al[e + 8] = selHi ? cll[e] : (_Float16)0.f;
        }
        // hi/lo split: ch*xh + ch*xl + cl*xh  (error ~2^-22, near-f32 accuracy)
        acc = __builtin_amdgcn_wmma_f32_16x16x32_f16(false, Al, false, Bh, (short)0, acc, false, false);
        acc = __builtin_amdgcn_wmma_f32_16x16x32_f16(false, Ah, false, Bl, (short)0, acc, false, false);
        acc = __builtin_amdgcn_wmma_f32_16x16x32_f16(false, Ah, false, Bh, (short)0, acc, false, false);
      }
      if (m < DDIM) {                              // D: VGPR r, lane -> (M = r + 8h, N = m)
#pragma unroll
        for (int r = 0; r < 8; ++r) {
          long gout = Gbase + tb + r + 8 * h;
          out[gout * DDIM + m] = acc[r];
        }
      }
    }
  } else {
    // ---------------- tail (< 32 groups): plain f32 per-lane path ----------------
    long g = Gbase + lane;
    if (g < (long)G) {
      const float* xp = x + g * (KBOX * DDIM);
      const float* cp = conf + g * KBOX;
      float wv[16], cvv[16];
      for (int i = 0; i < 16; ++i) { wv[i] = w[i]; cvv[i] = cp[i]; }
      int kcnt = 0;
      for (int k = 0; k < KBOX; ++k) {
        bool nz = false;
        for (int d = 0; d < DDIM; ++d) nz = nz || (xp[k * DDIM + d] != 0.f);
        kcnt += nz ? 1 : 0;
      }
      float wsum = 0.f, csum = 0.f;
      for (int k = 0; k < KBOX; ++k) if (k < kcnt) { wsum += wv[k]; csum += cvv[k]; }
      const float iw = 0.5f / wsum, ic = 0.5f / csum;
      float accd[DDIM];
      for (int d = 0; d < DDIM; ++d) accd[d] = 0.f;
      for (int k = 0; k < KBOX; ++k) {
        float coef = wv[k] * iw + cvv[k] * ic;
        for (int d = 0; d < DDIM; ++d) accd[d] += coef * xp[k * DDIM + d];
      }
      for (int d = 0; d < DDIM; ++d) out[g * DDIM + d] = accd[d];
    }
  }
}

extern "C" void kernel_launch(void* const* d_in, const int* in_sizes, int n_in,
                              void* d_out, int out_size, void* d_ws, size_t ws_size,
                              hipStream_t stream) {
  const float* x    = (const float*)d_in[0];
  const float* conf = (const float*)d_in[1];
  const float* w    = (const float*)d_in[2];
  float*       out  = (float*)d_out;
  const int G        = in_sizes[0] / (KBOX * DDIM);
  const int numWaves = (G + 31) / 32;
  const int blocks   = (numWaves + 1) / 2;
  weight_merge_kernel<<<blocks, 64, 0, stream>>>(x, conf, w, out, G);
}
